// MTRNN_4690104287703
// MI455X (gfx1250) — matrix operand order repaired and checked by reading, not compiled, not run on hardware
//
#include <hip/hip_runtime.h>

// ---------------------------------------------------------------------------
// MTRNN persistent kernel for MI455X (gfx1250), round 2.
//  - block-uniform roles (role = f(blockIdx.x)) -> scalar branches, no EXEC
//    masking around WMMA, specialized K-loops per role
//  - role-0 blocks stage I_t as bf16 into 64KB LDS with XOR bank swizzle;
//    inner-loop A fragments for the input GEMM come from ds_load_b128
//  - bf16 WMMA (v_wmma_f32_16x16x32_bf16) for all 9 GEMMs per step
//  - u-state in VGPRs; d double-buffered in L2-resident scratch
//  - 1 device-wide atomic barrier per timestep; readout pipelined 1 step late
// ---------------------------------------------------------------------------

typedef __attribute__((ext_vector_type(16))) __bf16 v16bf;
typedef __attribute__((ext_vector_type(8)))  __bf16 v8bf;
typedef __attribute__((ext_vector_type(8)))  float  v8f;
typedef __bf16 bf16_t;

#define TSTEPS 1024
#define MB     128
#define NIN    256
#define N0D    512
#define N1D    256
#define N2D    128
#define NOUTD  256
#define DTOT   896
#define NWG    18

// workspace layout (bytes), all offsets 256-aligned
#define OFF_BAR 0u
#define OFF_D0  256u
#define OFF_D1  229632u
#define OFF_W0  459008u       // Wcat0: 1024x512 bf16 (I|d0|d1 -> u0)
#define OFF_W1  1507584u      // Wcat1:  896x256 bf16 (d0|d1|d2 -> u1)
#define OFF_W2  1966336u      // Wcat2:  384x128 bf16 (d1|d2    -> u2)
#define OFF_WO  2064640u      // Wout :  512x256 bf16 (d0 -> out)

union Frag  { v16bf v; v8bf h[2]; uint4 q[2]; };
union Frag8 { v8bf  v; bf16_t e[8]; };

// B fragment (32x16 bf16): lane l holds column N = l%16, K rows 8g+s (s<8), 16+8g+(s-8) (s>=8).
__device__ __forceinline__ v16bf loadB(const bf16_t* wcol, int ld) {
  v16bf v;
#pragma unroll
  for (int s = 0; s < 8; ++s) {
    v[s] = wcol[(size_t)s * ld];
    v[s + 8] = wcol[(size_t)(16 + s) * ld];
  }
  return v;
}

__device__ __forceinline__ v8f wmma_bf16(v16bf a, v16bf b, v8f c) {
  return __builtin_amdgcn_wmma_f32_16x16x32_bf16(false, a, false, b,
                                                 (short)0, c, false, false);
}

__device__ __forceinline__ void grid_barrier(unsigned* cnt, unsigned* gen,
                                             unsigned* lg) {
  __syncthreads();
  if (threadIdx.x == 0) {
    unsigned old = __hip_atomic_fetch_add(cnt, 1u, __ATOMIC_ACQ_REL,
                                          __HIP_MEMORY_SCOPE_AGENT);
    if (old == NWG - 1u) {
      __hip_atomic_store(cnt, 0u, __ATOMIC_RELAXED, __HIP_MEMORY_SCOPE_AGENT);
      __hip_atomic_fetch_add(gen, 1u, __ATOMIC_RELEASE, __HIP_MEMORY_SCOPE_AGENT);
    } else {
      while (__hip_atomic_load(gen, __ATOMIC_ACQUIRE,
                               __HIP_MEMORY_SCOPE_AGENT) == *lg) {
        __builtin_amdgcn_s_sleep(2);
      }
    }
    ++*lg;
  }
  __syncthreads();
}

// LDS element index for I_t tile: row-major stride 256 with 16B XOR swizzle
// (k in units of elements, multiple of 8; swizzle bits = row's low 4 bits).
__device__ __forceinline__ int lds_idx(int r, int k) {
  return r * NIN + (k ^ ((r & 15) << 3));
}

// ---------------- layer role (u-update + tanh), templated --------------------
// KCI: K-chunks sourced from LDS-staged I_t; KCD: K-chunks from d buffer.
template <int KCI, int KCD, int LDN, int AOFF, int DOFF>
__device__ __forceinline__ void run_layer(
    int waveTask, int lane, float alpha,
    const float* __restrict__ I, bf16_t* ibuf,
    const bf16_t* __restrict__ Wm, const float* __restrict__ bias,
    bf16_t* dbuf0, bf16_t* dbuf1, unsigned* barCnt, unsigned* barGen)
{
  const int rg = waveTask & 1;
  const int c  = waveTask >> 1;
  const int g  = lane >> 4;
  const int lr = lane & 15;
  const int nbase = c * 16;
  const int rbase = rg * 64;
  const float bcol = bias[nbase + lr];

  v8f u_state[4] = {};
  unsigned localGen = 0;

  for (int t = 0; t <= TSTEPS; ++t) {
    const bf16_t* dR = (t & 1) ? dbuf1 : dbuf0;
    bf16_t*       dW = (t & 1) ? dbuf0 : dbuf1;

    if (t < TSTEPS) {
      if constexpr (KCI > 0) {
        // cooperative stage of I_t (fp32 -> bf16, swizzled) into LDS
        const float* Ib = I + (size_t)t * MB * NIN;
        const int tid = threadIdx.x;
#pragma unroll
        for (int p = 0; p < (MB * NIN) / (256 * 8); ++p) {
          const int base = p * 2048 + tid * 8;
          const int r = base >> 8, k0 = base & 255;
          const float* src = Ib + (size_t)r * NIN + k0;
          Frag8 f;
#pragma unroll
          for (int i = 0; i < 8; ++i) f.e[i] = (bf16_t)src[i];
          *(v8bf*)&ibuf[lds_idx(r, k0)] = f.v;
        }
        __syncthreads();
      }

      v8f acc[4] = {};
      const bf16_t* wrow = Wm + (size_t)(g * 8) * LDN + nbase + lr;

      if constexpr (KCI > 0) {
        for (int q = 0; q < KCI; ++q) {
          const v16bf bfrag = loadB(wrow, LDN);
          __builtin_prefetch(wrow + (size_t)32 * LDN, 0, 1);
          const int k0 = q * 32 + g * 8;
#pragma unroll
          for (int rt = 0; rt < 4; ++rt) {
            const int r = rbase + rt * 16 + lr;
            Frag f;
            f.h[0] = *(const v8bf*)&ibuf[lds_idx(r, k0)];
            f.h[1] = *(const v8bf*)&ibuf[lds_idx(r, k0 + 16)];
            acc[rt] = wmma_bf16(f.v, bfrag, acc[rt]);
          }
          wrow += (size_t)32 * LDN;
        }
      }

      const bf16_t* arow =
          dR + AOFF + (size_t)(rbase + lr) * DTOT + g * 8;
      for (int q = 0; q < KCD; ++q) {
        const v16bf bfrag = loadB(wrow, LDN);
        if (q + 1 < KCD) __builtin_prefetch(wrow + (size_t)32 * LDN, 0, 1);
#pragma unroll
        for (int rt = 0; rt < 4; ++rt) {
          const bf16_t* ap = arow + (size_t)rt * 16 * DTOT;
          Frag f;
          f.h[0] = *(const v8bf*)(ap);
          f.h[1] = *(const v8bf*)(ap + 16);
          acc[rt] = wmma_bf16(f.v, bfrag, acc[rt]);
        }
        arow += 32;
        wrow += (size_t)32 * LDN;
      }

      // u_new = (1-a)*u + a*(acc + b); d = tanh(u_new)
      // C/D layout: elem e -> row rbase+rt*16 + e + 8*g, col nbase+lr
#pragma unroll
      for (int rt = 0; rt < 4; ++rt) {
        const int r0 = rbase + rt * 16 + g * 8;
        bf16_t* dp = dW + (size_t)r0 * DTOT + DOFF + nbase + lr;
#pragma unroll
        for (int e = 0; e < 8; ++e) {
          float pre = acc[rt][e] + bcol;
          float un  = (1.f - alpha) * u_state[rt][e] + alpha * pre;
          u_state[rt][e] = un;
          dp[(size_t)e * DTOT] = (bf16_t)tanhf(un);
        }
      }
      grid_barrier(barCnt, barGen, &localGen);
    }
  }
}

// ---------------- readout role: out_{t-1} = d0(t-1) @ W_out + b_out ----------
__device__ __forceinline__ void run_out(
    int waveTask, int lane,
    const bf16_t* __restrict__ WO, const float* __restrict__ bo,
    const bf16_t* dbuf0, const bf16_t* dbuf1, float* __restrict__ out,
    unsigned* barCnt, unsigned* barGen)
{
  const int rg = waveTask & 1;
  const int c  = waveTask >> 1;
  const int g  = lane >> 4;
  const int lr = lane & 15;
  const int nbase = c * 16;
  const int rbase = rg * 64;
  const float bcol = bo[nbase + lr];
  unsigned localGen = 0;

  for (int t = 0; t <= TSTEPS; ++t) {
    const bf16_t* dR = (t & 1) ? dbuf1 : dbuf0;   // holds d(t-1)
    if (t >= 1) {
      v8f acc[4] = {};
      const bf16_t* wrow = WO + (size_t)(g * 8) * NOUTD + nbase + lr;
      const bf16_t* arow = dR + (size_t)(rbase + lr) * DTOT + g * 8;
      for (int q = 0; q < N0D / 32; ++q) {
        const v16bf bfrag = loadB(wrow, NOUTD);
        if (q + 1 < N0D / 32)
          __builtin_prefetch(wrow + (size_t)32 * NOUTD, 0, 1);
#pragma unroll
        for (int rt = 0; rt < 4; ++rt) {
          const bf16_t* ap = arow + (size_t)rt * 16 * DTOT;
          Frag f;
          f.h[0] = *(const v8bf*)(ap);
          f.h[1] = *(const v8bf*)(ap + 16);
          acc[rt] = wmma_bf16(f.v, bfrag, acc[rt]);
        }
        arow += 32;
        wrow += (size_t)32 * NOUTD;
      }
      float* op = out + (size_t)(t - 1) * MB * NOUTD;
#pragma unroll
      for (int rt = 0; rt < 4; ++rt) {
        const int r0 = rbase + rt * 16 + g * 8;
#pragma unroll
        for (int e = 0; e < 8; ++e)
          op[(size_t)(r0 + e) * NOUTD + nbase + lr] = acc[rt][e] + bcol;
      }
    }
    if (t < TSTEPS) grid_barrier(barCnt, barGen, &localGen);
  }
}

// ---------------- persistent kernel: role = f(blockIdx.x) -------------------
__global__ __launch_bounds__(256)
void mtrnn_persistent(const float* __restrict__ I,
                      const float* __restrict__ b0, const float* __restrict__ b1,
                      const float* __restrict__ b2, const float* __restrict__ bo,
                      const bf16_t* __restrict__ W0, const bf16_t* __restrict__ W1,
                      const bf16_t* __restrict__ W2, const bf16_t* __restrict__ WO,
                      bf16_t* dbuf0, bf16_t* dbuf1,
                      float* __restrict__ out,
                      unsigned* barCnt, unsigned* barGen)
{
  __shared__ bf16_t ibuf[MB * NIN];   // 64 KB: I_t staged as swizzled bf16
  const int lane = threadIdx.x & 31;
  const int wv   = threadIdx.x >> 5;
  const int b    = blockIdx.x;

  if (b < 8) {          // u0: 32 col-blocks x 2 row-groups = 64 wave-tasks
    run_layer<8, 24, N0D, 0, 0>(b * 8 + wv, lane, 0.5f, I, ibuf,
                                W0, b0, dbuf0, dbuf1, barCnt, barGen);
  } else if (b < 12) {  // u1: 32 wave-tasks
    run_layer<0, 28, N1D, 0, N0D>((b - 8) * 8 + wv, lane, 0.125f, nullptr,
                                  nullptr, W1, b1, dbuf0, dbuf1, barCnt, barGen);
  } else if (b < 14) {  // u2: 16 wave-tasks
    run_layer<0, 12, N2D, N0D, 768>((b - 12) * 8 + wv, lane, 0.03125f, nullptr,
                                    nullptr, W2, b2, dbuf0, dbuf1, barCnt, barGen);
  } else {              // readout: 32 wave-tasks
    run_out((b - 14) * 8 + wv, lane, WO, bo, dbuf0, dbuf1, out, barCnt, barGen);
  }
}

// ---------------------------- init kernels ---------------------------------

__global__ void zero_u32_kernel(unsigned* p, int n) {
  int i = blockIdx.x * blockDim.x + threadIdx.x;
  if (i < n) p[i] = 0u;
}

__global__ void f32_to_bf16_kernel(bf16_t* dst, const float* src, int n) {
  int i = blockIdx.x * blockDim.x + threadIdx.x;
  if (i < n) dst[i] = (bf16_t)src[i];
}

// ---------------------------- launcher -------------------------------------

extern "C" void kernel_launch(void* const* d_in, const int* in_sizes, int n_in,
                              void* d_out, int out_size, void* d_ws, size_t ws_size,
                              hipStream_t stream) {
  const float* I      = (const float*)d_in[0];
  const float* W0_low = (const float*)d_in[1];
  const float* W0_rec = (const float*)d_in[2];
  const float* W0_hi  = (const float*)d_in[3];
  const float* b0     = (const float*)d_in[4];
  const float* W1_low = (const float*)d_in[5];
  const float* W1_rec = (const float*)d_in[6];
  const float* W1_hi  = (const float*)d_in[7];
  const float* b1     = (const float*)d_in[8];
  const float* W2_low = (const float*)d_in[9];
  const float* W2_rec = (const float*)d_in[10];
  const float* b2     = (const float*)d_in[11];
  const float* W_out  = (const float*)d_in[12];
  const float* b_out  = (const float*)d_in[13];
  float* out = (float*)d_out;

  char* ws = (char*)d_ws;
  unsigned* bar   = (unsigned*)(ws + OFF_BAR);
  bf16_t*   dbuf0 = (bf16_t*)(ws + OFF_D0);
  bf16_t*   dbuf1 = (bf16_t*)(ws + OFF_D1);
  bf16_t*   W0b   = (bf16_t*)(ws + OFF_W0);
  bf16_t*   W1b   = (bf16_t*)(ws + OFF_W1);
  bf16_t*   W2b   = (bf16_t*)(ws + OFF_W2);
  bf16_t*   WOb   = (bf16_t*)(ws + OFF_WO);

  {
    int zn = (int)(OFF_W0 / 4);   // barrier state + both d buffers
    zero_u32_kernel<<<(zn + 255) / 256, 256, 0, stream>>>((unsigned*)ws, zn);
  }

  auto cvt = [&](bf16_t* dst, const float* src, int n) {
    f32_to_bf16_kernel<<<(n + 255) / 256, 256, 0, stream>>>(dst, src, n);
  };
  cvt(W0b,                     W0_low, NIN * N0D);
  cvt(W0b + NIN * N0D,         W0_rec, N0D * N0D);
  cvt(W0b + (NIN + N0D) * N0D, W0_hi,  N1D * N0D);
  cvt(W1b,                     W1_low, N0D * N1D);
  cvt(W1b + N0D * N1D,         W1_rec, N1D * N1D);
  cvt(W1b + (N0D + N1D) * N1D, W1_hi,  N2D * N1D);
  cvt(W2b,                     W2_low, N1D * N2D);
  cvt(W2b + N1D * N2D,         W2_rec, N2D * N2D);
  cvt(WOb,                     W_out,  N0D * NOUTD);

  mtrnn_persistent<<<NWG, 256, 0, stream>>>(I, b0, b1, b2, b_out,
                                            W0b, W1b, W2b, WOb,
                                            dbuf0, dbuf1, out,
                                            bar, bar + 1);
}
